// LASDecoder_22368189677708
// MI455X (gfx1250) — matrix operand order repaired and checked by reading, not compiled, not run on hardware
//
#include <hip/hip_runtime.h>
#include <hip/hip_bf16.h>
#include <math.h>

// Problem constants (from reference)
#define BSZ   16
#define KLN   512
#define TT    96
#define DIM   512
#define VOC   10000
#define CKN   201
#define CON   10

typedef __attribute__((ext_vector_type(16))) _Float16 v16h;
typedef __attribute__((ext_vector_type(8)))  _Float16 v8h;
typedef __attribute__((ext_vector_type(8)))  float    v8f;

union ABFrag { v16h v; v8h h[2]; };

__device__ __forceinline__ v8f wmma_f16(const ABFrag& a, const ABFrag& b, v8f c) {
  return __builtin_amdgcn_wmma_f32_16x16x32_f16(
      false, a.v, false, b.v, (short)0, c, false, false);
}

// ---------------------------------------------------------------------------
// Core WMMA helper: acc(16x16) += X(16 x K) * W^T  for output columns
// [n0, n0+16).  X rows may be a concatenation of two sources split at `split`
// (split is a multiple of 32 so 32-wide K blocks never straddle it).
// A-fragment layout (16-bit A 16x32, wave32):
//   lanes 0-15  : K in [k0, k0+8) and [k0+16, k0+24)
//   lanes 16-31 : K in [k0+8, k0+16) and [k0+24, k0+32)
// B-fragment layout (32x16): lane L holds column n = L%16, same K ranges
//   as A: [k0, k0+8)+[k0+16, k0+24) for lanes 0-15, [k0+8, k0+16)+
//   [k0+24, k0+32) for lanes 16-31 — slices of row n of W (B = W^T).
// ---------------------------------------------------------------------------
__device__ __forceinline__ void wmma_gemm_acc(v8f& acc,
    const _Float16* __restrict__ X0, int ld0,
    const _Float16* __restrict__ X1, int ld1, int split,
    const _Float16* __restrict__ W, int K, int n0)
{
  const int lane = threadIdx.x & 31;
  const int m    = lane & 15;
  const int hi   = lane >> 4;
  const _Float16* wrow = W + (size_t)(n0 + m) * (size_t)K;
  for (int k0 = 0; k0 < K; k0 += 32) {
    const _Float16* xs = (k0 < split) ? (X0 + (size_t)m * ld0 + k0)
                                      : (X1 + (size_t)m * ld1 + (k0 - split));
    ABFrag a, b;
    a.h[0] = *(const v8h*)(xs + hi * 8);
    a.h[1] = *(const v8h*)(xs + 16 + hi * 8);
    const _Float16* wp = wrow + k0 + hi * 8;
    b.h[0] = *(const v8h*)(wp);
    b.h[1] = *(const v8h*)(wp + 16);
    acc = wmma_f16(a, b, acc);
  }
}

// ---------------------------------------------------------------------------
// Generic single-tile WMMA GEMM (for the small latency-bound per-step GEMMs):
// out = act(X @ W^T + bias).  One 16x16 tile per wave.
// D layout: acc[j] -> row (mt*16 + j + 8*hi), col (nt*16 + lane%16).
// ---------------------------------------------------------------------------
__global__ void gemm_kernel(const _Float16* __restrict__ X0, int ld0,
                            const _Float16* __restrict__ X1, int ld1, int split,
                            const _Float16* __restrict__ W, int K,
                            const float* __restrict__ bias,
                            float* outF, _Float16* outH, long long rowStride,
                            int Ntiles, int tiles, int act)
{
  int gwave = (int)((blockIdx.x * blockDim.x + threadIdx.x) >> 5);
  if (gwave >= tiles) return;
  int mt = gwave / Ntiles, nt = gwave % Ntiles;
  v8f acc = {};
  wmma_gemm_acc(acc, X0 + (size_t)mt * 16 * ld0, ld0,
                     X1 + (size_t)mt * 16 * ld1, ld1, split, W, K, nt * 16);
  int lane = threadIdx.x & 31, m = lane & 15, hi = lane >> 4;
  int n = nt * 16 + m;
  float bv = bias[n];
#pragma unroll
  for (int j = 0; j < 8; j++) {
    int row = mt * 16 + j + 8 * hi;
    float v = acc[j] + bv;
    if (act) v = tanhf(v);
    if (outF) outF[(size_t)row * rowStride + n] = v;
    else      outH[(size_t)row * rowStride + n] = (_Float16)v;
  }
}

// ---------------------------------------------------------------------------
// 2x2 register-tiled WMMA GEMM for the big throughput GEMMs (key projection,
// output projection).  Each wave computes a 32x32 output block with 4
// accumulators; A fragments are reused across the two N-tiles and B fragments
// across the two M-tiles -> operand traffic per WMMA is halved (8 -> 16
// FLOP/byte), which matters because these GEMMs are L2-bandwidth-bound.
// Edge guards are wave-uniform (EXEC stays all-ones as WMMA requires).
// ---------------------------------------------------------------------------
__global__ void gemm2x2_kernel(const _Float16* __restrict__ X, int ldx,
                               const _Float16* __restrict__ W, int K,
                               const float* __restrict__ bias,
                               float* __restrict__ outF, long long rowStride,
                               int Mtiles, int Ntiles, int act)
{
  const int Nb = (Ntiles + 1) >> 1;
  const int Mb = (Mtiles + 1) >> 1;
  int wave = (int)((blockIdx.x * blockDim.x + threadIdx.x) >> 5);
  if (wave >= Mb * Nb) return;
  const int mb = wave / Nb, nb = wave % Nb;
  const int mt0 = mb * 2, nt0 = nb * 2;
  const bool hasM1 = (mt0 + 1) < Mtiles;
  const bool hasN1 = (nt0 + 1) < Ntiles;

  const int lane = threadIdx.x & 31;
  const int m = lane & 15, hi = lane >> 4;
  const _Float16* a0p = X + (size_t)(mt0 * 16 + m) * ldx;
  const _Float16* a1p = a0p + (size_t)16 * ldx;
  const _Float16* b0p = W + (size_t)(nt0 * 16 + m) * K;
  const _Float16* b1p = b0p + (size_t)16 * K;

  v8f acc00 = {}, acc01 = {}, acc10 = {}, acc11 = {};
  for (int k0 = 0; k0 < K; k0 += 32) {
    ABFrag a0, a1, b0, b1;
    a0.h[0] = *(const v8h*)(a0p + k0 + hi * 8);
    a0.h[1] = *(const v8h*)(a0p + k0 + 16 + hi * 8);
    b0.h[0] = *(const v8h*)(b0p + k0 + hi * 8);
    b0.h[1] = *(const v8h*)(b0p + k0 + 16 + hi * 8);
    acc00 = wmma_f16(a0, b0, acc00);
    if (hasM1) {
      a1.h[0] = *(const v8h*)(a1p + k0 + hi * 8);
      a1.h[1] = *(const v8h*)(a1p + k0 + 16 + hi * 8);
      acc10 = wmma_f16(a1, b0, acc10);
    }
    if (hasN1) {
      b1.h[0] = *(const v8h*)(b1p + k0 + hi * 8);
      b1.h[1] = *(const v8h*)(b1p + k0 + 16 + hi * 8);
      acc01 = wmma_f16(a0, b1, acc01);
      if (hasM1) acc11 = wmma_f16(a1, b1, acc11);
    }
  }

  auto store_tile = [&](const v8f& acc, int mt, int nt) {
    int n = nt * 16 + m;
    float bv = bias[n];
#pragma unroll
    for (int j = 0; j < 8; j++) {
      int row = mt * 16 + j + 8 * hi;
      float v = acc[j] + bv;
      if (act) v = tanhf(v);
      outF[(size_t)row * rowStride + n] = v;
    }
  };
  store_tile(acc00, mt0, nt0);
  if (hasM1) store_tile(acc10, mt0 + 1, nt0);
  if (hasN1) {
    store_tile(acc01, mt0, nt0 + 1);
    if (hasM1) store_tile(acc11, mt0 + 1, nt0 + 1);
  }
}

// ---------------------------------------------------------------------------
// Fused LSTM step: gates = [X] @ W_ih^T + h_prev @ W_hh^T + b_ih + b_hh,
// then cell update.  Grid = 32 blocks (16 gate columns each), block = 128
// (4 waves; wave s computes the tile of gate section s: i/f/g/o).
// Gates staged through LDS so the elementwise cell update sees all four.
// h written to a double buffer (hNew != hPrev) to avoid cross-WG RAW.
// ---------------------------------------------------------------------------
__global__ void lstm_kernel(const _Float16* __restrict__ X0, int ld0,
                            const _Float16* __restrict__ X1, int ld1,
                            int split, int Kin,
                            const _Float16* __restrict__ hPrev,
                            const _Float16* __restrict__ Wih,
                            const _Float16* __restrict__ Whh,
                            const float* __restrict__ bih,
                            const float* __restrict__ bhh,
                            float* __restrict__ cBuf,
                            _Float16* __restrict__ hNew)
{
  __shared__ float g[4][16][17];
  const int waveId = threadIdx.x >> 5;           // gate section 0..3 (i,f,g,o)
  const int n0 = blockIdx.x * 16;                // columns within a section
  const int gn0 = waveId * DIM + n0;             // global gate column base
  v8f acc = {};
  wmma_gemm_acc(acc, X0, ld0, X1, ld1, split, Wih, Kin, gn0);
  wmma_gemm_acc(acc, hPrev, DIM, hPrev, DIM, DIM, Whh, DIM, gn0);
  int lane = threadIdx.x & 31, m = lane & 15, hi = lane >> 4;
  float bv = bih[gn0 + m] + bhh[gn0 + m];
#pragma unroll
  for (int j = 0; j < 8; j++) g[waveId][j + 8 * hi][m] = acc[j] + bv;
  __syncthreads();
  for (int e = threadIdx.x; e < 256; e += 128) {
    int r = e >> 4, c = e & 15;                  // r = batch row, c = column
    int idx = r * DIM + n0 + c;
    float iv = g[0][r][c], fv = g[1][r][c], gv = g[2][r][c], ov = g[3][r][c];
    float si = 1.f / (1.f + __expf(-iv));
    float sf = 1.f / (1.f + __expf(-fv));
    float so = 1.f / (1.f + __expf(-ov));
    float cn = sf * cBuf[idx] + si * tanhf(gv);
    cBuf[idx] = cn;
    hNew[idx] = (_Float16)(so * tanhf(cn));
  }
}

// ---------------------------------------------------------------------------
// Attention: one workgroup per batch element.  conv(aw) -> energies with
// tanh(key_proj + q_proj + c_proj)·ws -> masked softmax -> context.
// aw/energies/conv features all LDS-resident; key_proj/eouts stream from L2.
// ---------------------------------------------------------------------------
__global__ void att_kernel(const float* __restrict__ keyproj,
                           const float* __restrict__ qproj,
                           const float* __restrict__ convw,
                           const float* __restrict__ Wc,
                           const float* __restrict__ bc,
                           const float* __restrict__ ws_,
                           const float* __restrict__ bs_,
                           const int*   __restrict__ elens,
                           const float* __restrict__ eouts,
                           float* __restrict__ aw,
                           float* __restrict__ ctx,
                           _Float16* __restrict__ ctxH)
{
  const int b = blockIdx.x;
  const int tid = threadIdx.x;
  __shared__ float aw_s[KLN];
  __shared__ float cf[KLN][CON];
  __shared__ float e_s[KLN];
  __shared__ float red[256];

  for (int k = tid; k < KLN; k += 256) aw_s[k] = aw[b * KLN + k];
  __syncthreads();

  // 1D conv of previous attention weights (pad = 100)
  for (int o = tid; o < KLN * CON; o += 256) {
    int k = o / CON, co = o % CON;
    float s = 0.f;
    for (int ck = 0; ck < CKN; ck++) {
      int p = k - 100 + ck;
      if (p >= 0 && p < KLN) s += aw_s[p] * convw[co * CKN + ck];
    }
    cf[k][co] = s;
  }
  __syncthreads();

  // Energies: one wave per key index, lanes reduce over feature dim.
  int wave = tid >> 5, lane = tid & 31;
  int klen = elens[b];
  const float* kp = keyproj + (size_t)b * KLN * DIM;
  const float* qp = qproj + b * DIM;
  for (int k = wave; k < KLN; k += 8) {
    float part = 0.f;
    for (int d = lane; d < DIM; d += 32) {
      float cp = bc[d];
#pragma unroll
      for (int co = 0; co < CON; co++) cp += cf[k][co] * Wc[d * CON + co];
      part += tanhf(kp[(size_t)k * DIM + d] + qp[d] + cp) * ws_[d];
    }
    for (int off = 16; off > 0; off >>= 1) part += __shfl_down(part, off);
    if (lane == 0)
      e_s[k] = (k < klen) ? (part + bs_[0]) : -__FLT_MAX__;
  }
  __syncthreads();

  // Softmax over the 512 keys
  float lmax = -__FLT_MAX__;
  for (int k = tid; k < KLN; k += 256) lmax = fmaxf(lmax, e_s[k]);
  red[tid] = lmax; __syncthreads();
  for (int s = 128; s > 0; s >>= 1) {
    if (tid < s) red[tid] = fmaxf(red[tid], red[tid + s]);
    __syncthreads();
  }
  float mx = red[0];
  __syncthreads();
  float lsum = 0.f;
  for (int k = tid; k < KLN; k += 256) {
    float v = __expf(e_s[k] - mx);
    e_s[k] = v; lsum += v;
  }
  red[tid] = lsum; __syncthreads();
  for (int s = 128; s > 0; s >>= 1) {
    if (tid < s) red[tid] += red[tid + s];
    __syncthreads();
  }
  float inv = 1.f / red[0];
  __syncthreads();
  for (int k = tid; k < KLN; k += 256) {
    float v = e_s[k] * inv;
    aw[b * KLN + k] = v;
    aw_s[k] = v;
  }
  __syncthreads();

  // Context vector: ctx[d] = sum_k aw[k] * eouts[b,k,d]
  const float* eo = eouts + (size_t)b * KLN * DIM;
  for (int d = tid; d < DIM; d += 256) {
    float s = 0.f;
    for (int k = 0; k < KLN; k++) s += aw_s[k] * eo[(size_t)k * DIM + d];
    ctx[b * DIM + d] = s;
    ctxH[b * DIM + d] = (_Float16)s;
  }
}

// ---------------------------------------------------------------------------
// Utility kernels
// ---------------------------------------------------------------------------
__global__ void f2h_kernel(const float* __restrict__ in, _Float16* out, int n) {
  int stride = gridDim.x * blockDim.x;
  for (int i = blockIdx.x * blockDim.x + threadIdx.x; i < n; i += stride)
    out[i] = (_Float16)in[i];
}

__global__ void emb_kernel(const float* __restrict__ embed,
                           const int* __restrict__ ys, _Float16* out, int n) {
  int stride = gridDim.x * blockDim.x;
  for (int i = blockIdx.x * blockDim.x + threadIdx.x; i < n; i += stride) {
    int row = i >> 9, d = i & 511;
    out[i] = (_Float16)embed[(size_t)ys[row] * DIM + d];
  }
}

__global__ void zero_kernel(unsigned int* p, int nwords) {
  int stride = gridDim.x * blockDim.x;
  for (int i = blockIdx.x * blockDim.x + threadIdx.x; i < nwords; i += stride)
    p[i] = 0u;
}

// ---------------------------------------------------------------------------
// Host orchestration
// ---------------------------------------------------------------------------
extern "C" void kernel_launch(void* const* d_in, const int* in_sizes, int n_in,
                              void* d_out, int out_size, void* d_ws, size_t ws_size,
                              hipStream_t stream) {
  // Input order: eouts, params{embed, att_Wk, att_bk, att_Wq, att_bq, att_Wc,
  // att_bc, att_ws, att_bs, conv_w, inter_W, inter_b, out_W, out_b,
  // W_ih_0, W_hh_0, b_ih_0, b_hh_0, W_ih_1, W_hh_1, b_ih_1, b_hh_1}, elens, ys_in
  const float* eouts  = (const float*)d_in[0];
  const float* embed  = (const float*)d_in[1];
  const float* Wk     = (const float*)d_in[2];
  const float* bk     = (const float*)d_in[3];
  const float* Wq     = (const float*)d_in[4];
  const float* bq     = (const float*)d_in[5];
  const float* Wc     = (const float*)d_in[6];
  const float* bc     = (const float*)d_in[7];
  const float* wsv    = (const float*)d_in[8];
  const float* bsv    = (const float*)d_in[9];
  const float* convw  = (const float*)d_in[10];
  const float* interW = (const float*)d_in[11];
  const float* interB = (const float*)d_in[12];
  const float* outW   = (const float*)d_in[13];
  const float* outB   = (const float*)d_in[14];
  const float* Wih0   = (const float*)d_in[15];
  const float* Whh0   = (const float*)d_in[16];
  const float* bih0   = (const float*)d_in[17];
  const float* bhh0   = (const float*)d_in[18];
  const float* Wih1   = (const float*)d_in[19];
  const float* Whh1   = (const float*)d_in[20];
  const float* bih1   = (const float*)d_in[21];
  const float* bhh1   = (const float*)d_in[22];
  const int*   elens  = (const int*)d_in[23];
  const int*   ys_in  = (const int*)d_in[24];

  // Bump allocator over d_ws (256B aligned)
  char* p = (char*)d_ws;
  auto alloc = [&](size_t bytes) -> char* {
    char* r = p; p += (bytes + 255) & ~(size_t)255; return r;
  };

  // Recurrent state block (zeroed every call, contiguous)
  char* stateStart = p;
  _Float16* h0h[2] = { (_Float16*)alloc(BSZ * DIM * 2), (_Float16*)alloc(BSZ * DIM * 2) };
  _Float16* h1h[2] = { (_Float16*)alloc(BSZ * DIM * 2), (_Float16*)alloc(BSZ * DIM * 2) };
  _Float16* ctxH   = (_Float16*)alloc(BSZ * DIM * 2);
  float* c0    = (float*)alloc(BSZ * DIM * 4);
  float* c1    = (float*)alloc(BSZ * DIM * 4);
  float* ctx   = (float*)alloc(BSZ * DIM * 4);
  float* aw    = (float*)alloc(BSZ * KLN * 4);
  float* qproj = (float*)alloc(BSZ * DIM * 4);
  size_t stateBytes = (size_t)(p - stateStart);

  // Persistent (fully rewritten every call) scratch
  _Float16* eoutsH  = (_Float16*)alloc((size_t)BSZ * KLN * DIM * 2);
  _Float16* wkH     = (_Float16*)alloc((size_t)DIM * DIM * 2);
  _Float16* wqH     = (_Float16*)alloc((size_t)DIM * DIM * 2);
  _Float16* winterH = (_Float16*)alloc((size_t)DIM * 2 * DIM * 2);
  _Float16* woutH   = (_Float16*)alloc((size_t)VOC * DIM * 2);
  _Float16* wih0H   = (_Float16*)alloc((size_t)4 * DIM * 2 * DIM * 2);
  _Float16* whh0H   = (_Float16*)alloc((size_t)4 * DIM * DIM * 2);
  _Float16* wih1H   = (_Float16*)alloc((size_t)4 * DIM * DIM * 2);
  _Float16* whh1H   = (_Float16*)alloc((size_t)4 * DIM * DIM * 2);
  _Float16* yembH   = (_Float16*)alloc((size_t)BSZ * TT * DIM * 2);
  _Float16* logitsH = (_Float16*)alloc((size_t)BSZ * TT * DIM * 2);
  float*    keyproj = (float*)alloc((size_t)BSZ * KLN * DIM * 4);

  // --- Setup: zero state, convert weights/activations to f16, gather emb ---
  zero_kernel<<<256, 256, 0, stream>>>((unsigned int*)stateStart, (int)(stateBytes / 4));

  auto cvt = [&](const float* src, _Float16* dst, int n) {
    int blocks = (n + 255) / 256; if (blocks > 4096) blocks = 4096;
    f2h_kernel<<<blocks, 256, 0, stream>>>(src, dst, n);
  };
  cvt(eouts,  eoutsH,  BSZ * KLN * DIM);
  cvt(Wk,     wkH,     DIM * DIM);
  cvt(Wq,     wqH,     DIM * DIM);
  cvt(interW, winterH, DIM * 2 * DIM);
  cvt(outW,   woutH,   VOC * DIM);
  cvt(Wih0,   wih0H,   4 * DIM * 2 * DIM);
  cvt(Whh0,   whh0H,   4 * DIM * DIM);
  cvt(Wih1,   wih1H,   4 * DIM * DIM);
  cvt(Whh1,   whh1H,   4 * DIM * DIM);
  emb_kernel<<<4096, 256, 0, stream>>>(embed, ys_in, yembH, BSZ * TT * DIM);

  // key_proj = eouts @ Wk^T + bk : M = 8192 rows, N = 512, K = 512
  // 2x2 tiled: Mb = 256, Nb = 16 -> 4096 waves -> 512 blocks
  gemm2x2_kernel<<<512, 256, 0, stream>>>(eoutsH, DIM, wkH, DIM, bk,
      keyproj, (long long)DIM, /*Mtiles=*/BSZ * KLN / 16, /*Ntiles=*/32, 0);

  // --- Recurrent decode: 96 steps ---
  for (int t = 0; t < TT; t++) {
    const _Float16* h0p = h0h[t & 1];       _Float16* h0n = h0h[(t + 1) & 1];
    const _Float16* h1p = h1h[t & 1];       _Float16* h1n = h1h[(t + 1) & 1];

    // LSTM layer 0: X = [y_emb_t (512) | ctx (512)], K = 1024
    lstm_kernel<<<32, 128, 0, stream>>>(
        yembH + (size_t)t * DIM, TT * DIM, ctxH, DIM, DIM, 2 * DIM,
        h0p, wih0H, whh0H, bih0, bhh0, c0, h0n);

    // q_proj = h0_new @ Wq^T + bq : M = 16, N = 512, K = 512
    gemm_kernel<<<4, 256, 0, stream>>>(h0n, DIM, h0n, DIM, DIM,
        wqH, DIM, bq, qproj, (_Float16*)nullptr, (long long)DIM, 32, 32, 0);

    // LSTM layer 1: X = h0_new, K = 512
    lstm_kernel<<<32, 128, 0, stream>>>(
        h0n, DIM, h0n, DIM, DIM, DIM,
        h1p, wih1H, whh1H, bih1, bhh1, c1, h1n);

    // Location-aware attention (updates aw, ctx, ctxH)
    att_kernel<<<BSZ, 256, 0, stream>>>(keyproj, qproj, convw, Wc, bc,
        wsv, bsv, elens, eouts, aw, ctx, ctxH);

    // logit_t = tanh([ctx_new | h1_new] @ inter_W^T + b) -> logits_f16[b][t][:]
    gemm_kernel<<<4, 256, 0, stream>>>(ctxH, DIM, h1n, DIM, DIM,
        winterH, 2 * DIM, interB, (float*)nullptr, logitsH + (size_t)t * DIM,
        (long long)TT * DIM, 32, 32, /*act=*/1);
  }

  // --- Output projection: (16*96, 512) @ (10000, 512)^T + out_b ---
  // 2x2 tiled: Mb = 48, Nb = 313 (N edge handled) -> 15024 waves -> 1878 blocks
  gemm2x2_kernel<<<1878, 256, 0, stream>>>(logitsH, DIM, woutH, DIM, outB,
      (float*)d_out, (long long)VOC, /*Mtiles=*/96, /*Ntiles=*/625, 0);
}
